// MultimodalModel_32384053411895
// MI455X (gfx1250) — matrix operand order repaired and checked by reading, not compiled, not run on hardware
//
#include <hip/hip_runtime.h>
#include <hip/hip_bf16.h>
#include <math.h>

// ---------------------------------------------------------------------------
// MI455X (gfx1250) implementation of the multimodal transformer reference.
// All large GEMMs use V_WMMA_F32_16X16X32_BF16 (bf16 in, f32 accumulate),
// with weight tiles staged into LDS through the CDNA5 async global->LDS
// path (GLOBAL_LOAD_ASYNC_TO_LDS_B128 / ASYNCcnt), double buffered so the
// DMA overlaps the WMMA stream. B fragments are software-pipelined one
// K-step ahead so ds_load latency is hidden under the WMMAs.
// ---------------------------------------------------------------------------

typedef __bf16 bf16;
typedef bf16  v16bf __attribute__((ext_vector_type(16)));
typedef bf16  v8bf  __attribute__((ext_vector_type(8)));
typedef float v8f   __attribute__((ext_vector_type(8)));

#define T_SEQ  256
#define BATCH  64
#define DMODEL 512
#define NTOK   (T_SEQ * BATCH)   // 16384 tokens, row-major (t*64+b, 512)

// ---- CDNA5 async copy helpers ---------------------------------------------
// Per-lane: LDS[ldsAddr] = MEM[gaddr] (16 bytes), tracked by ASYNCcnt.
__device__ __forceinline__ void async_ld_b128(unsigned ldsAddr, const bf16* gaddr) {
  asm volatile("global_load_async_to_lds_b128 %0, %1, off"
               :: "v"(ldsAddr), "v"(gaddr) : "memory");
}
__device__ __forceinline__ void wait_asynccnt_0() {
  asm volatile("s_wait_asynccnt 0x0" ::: "memory");
}
__device__ __forceinline__ void wait_asynccnt_8() {
  asm volatile("s_wait_asynccnt 0x8" ::: "memory");
}

// ---- WMMA fragment loaders -------------------------------------------------
// A-matrix (16x32 bf16, M x K): lanes 0-15 hold M=lane, K=[0..15];
// lanes 16-31 hold M=lane-16, K=[16..31] (contiguous, matches B loader).
__device__ __forceinline__ v16bf load_a_frag(const bf16* A, int lda,
                                             int row0, int k0, int lane) {
  const int m  = lane & 15;
  const int kb = (lane >> 4) << 4;           // 0 or 16
  const bf16* p = A + (size_t)(row0 + m) * lda + (k0 + kb);
  union { v16bf v; v8bf h[2]; } f;
  f.h[0] = *(const v8bf*)(p);                // K = k0+kb .. +7   (16B load)
  f.h[1] = *(const v8bf*)(p + 8);            // K = k0+kb+8 .. +15
  return f.v;
}

// B-matrix (32x16 bf16, K x N) with B[k][n] = W[n][k], W row-major (N,K):
// lane n=lane&15 is the column; VGPR v packs K=2v,2v+1 (lanes<16) or +16
// (lanes>=16)  => one contiguous 32-byte load from W row (col0+n).
__device__ __forceinline__ v16bf load_b_frag(const bf16* W, int ldw,
                                             int col0, int k0, int lane) {
  const int n  = lane & 15;
  const int ko = (lane >> 4) << 4;           // 0 or 16
  return *(const v16bf*)(W + (size_t)(col0 + n) * ldw + (k0 + ko));
}

__device__ __forceinline__ v8f wmma_bf16(v16bf a, v16bf b, v8f c) {
  return __builtin_amdgcn_wmma_f32_16x16x32_bf16(false, a, false, b,
                                                 (short)0, c, false, false);
}

// ---- Generic GEMM: out = act(A[M,K] * W[N,K]^T + bias) (+ residual) --------
// block = 128 threads (4 waves); each wave: 16 rows x 64 cols; grid(N/64, M/64).
// Weight tile (64 cols x 128 K) is staged in LDS via async global->LDS copies,
// double buffered: the DMA for chunk i+1 overlaps the WMMAs of chunk i.
#define KC 128
template <bool RELU>
__global__ void gemm_bf16(const bf16* __restrict__ A, const bf16* __restrict__ W,
                          const float* __restrict__ bias,
                          const float* __restrict__ residual,
                          float* __restrict__ outF, bf16* __restrict__ outB,
                          int M, int N, int K) {
  __shared__ bf16 Wlds[2][64][KC];           // 2 x 16 KB
  const int tid   = threadIdx.x;
  const int lane  = tid & 31;
  const int wave  = tid >> 5;
  const int tileN = blockIdx.x * 64;
  const int tileM = blockIdx.y * 64 + wave * 16;
  // On CDNA5 the low 32 bits of a generic LDS pointer are the LDS address.
  const unsigned ldsBase = (unsigned)(size_t)(&Wlds[0][0][0]);

  // One chunk = 64 rows x 128 K-elems = 16 KB = 8 async b128 issues per wave.
  auto issue_chunk = [&](int buf, int k0) {
#pragma unroll
    for (int q = 0; q < 8; ++q) {
      const int e  = (q * 128 + tid) * 8;    // bf16 element index in chunk
      const int n  = e >> 7;                 // row (= W output col) 0..63
      const int kk = e & (KC - 1);
      async_ld_b128(ldsBase + (unsigned)buf * (64 * KC * 2) + (unsigned)e * 2,
                    W + (size_t)(tileN + n) * K + (k0 + kk));
    }
  };

  v8f acc[4] = {};
  const int nch = K >> 7;                    // K / 128 (K in {512,1536,2048})
  issue_chunk(0, 0);
  for (int ch = 0; ch < nch; ++ch) {
    if (ch + 1 < nch) {
      issue_chunk((ch + 1) & 1, (ch + 1) << 7);
      wait_asynccnt_8();                     // 8 older issues (chunk ch) done
    } else {
      wait_asynccnt_0();
    }
    __syncthreads();                         // all waves' copies visible
    const bf16* Wc = &Wlds[ch & 1][0][0];

    // Software-pipeline the B fragments one K-step ahead: both fragment sets
    // are live simultaneously, so the ds_loads for step kk+32 issue before
    // the four WMMAs of step kk and their latency is hidden by XDL work.
    v16bf bfr[2][4];
#pragma unroll
    for (int c = 0; c < 4; ++c)
      bfr[0][c] = load_b_frag(Wc, KC, c * 16, 0, lane);
#pragma unroll
    for (int kk = 0; kk < KC; kk += 32) {
      const int cu = (kk >> 5) & 1;
      v16bf a = load_a_frag(A, K, tileM, (ch << 7) + kk, lane);
      if (kk + 32 < KC) {
#pragma unroll
        for (int c = 0; c < 4; ++c)
          bfr[cu ^ 1][c] = load_b_frag(Wc, KC, c * 16, kk + 32, lane);
      }
#pragma unroll
      for (int c = 0; c < 4; ++c)
        acc[c] = wmma_bf16(a, bfr[cu][c], acc[c]);
    }
    __syncthreads();                         // buffer may be overwritten next
  }

  // C layout: lane&15 = N col in tile; VGPR r -> M = r (+8 for lanes>=16)
  const int ncol = lane & 15;
  const int lh   = lane >> 4;
#pragma unroll
  for (int c = 0; c < 4; ++c) {
    const int col = tileN + c * 16 + ncol;
    const float bv = bias ? bias[col] : 0.f;
#pragma unroll
    for (int r = 0; r < 8; ++r) {
      const int row = tileM + lh * 8 + r;
      float v = acc[c][r] + bv;
      if (RELU) v = fmaxf(v, 0.f);
      const size_t idx = (size_t)row * N + col;
      if (residual) v += residual[idx];
      if (outF) outF[idx] = v;
      if (outB) outB[idx] = (bf16)v;
    }
  }
}

// ---- Attention over "pairs": pair p -> (t = p/H, h = p%H) ------------------
// Q,K,V tiles are 64 rows x DK cols at base t*64*ld + h*DK (bf16, row major).
// block = 128 threads (4 waves); wave w owns S/ctx rows 16w..16w+15.
template <int DK>
__global__ void attn_kernel(const bf16* __restrict__ Q, const bf16* __restrict__ K,
                            const bf16* __restrict__ V, bf16* __restrict__ CTX,
                            int H, int ld, int ldo, float scale) {
  __shared__ bf16 Pl[4][16][64];  // softmaxed probabilities, A-frag layout
  __shared__ bf16 Vt[DK][64];     // V transposed -> B-frag friendly
  const int p = blockIdx.x;
  const int t = p / H, h = p % H;
  const size_t qb = (size_t)t * 64 * ld  + (size_t)h * DK;
  const size_t ob = (size_t)t * 64 * ldo + (size_t)h * DK;
  const int lane = threadIdx.x & 31;
  const int wave = threadIdx.x >> 5;

  // Cooperative transpose of the V tile into LDS.
  for (int i = threadIdx.x; i < 64 * DK; i += 128) {
    const int j = i / DK, dd = i % DK;
    Vt[dd][j] = V[qb + (size_t)j * ld + dd];
  }

  // S = (Q * K^T) * scale   (K rows are contiguous -> same loader as W^T)
  v8f s[4] = {};
  for (int k0 = 0; k0 < DK; k0 += 32) {
    v16bf a = load_a_frag(Q + qb, ld, wave * 16, k0, lane);
    v16bf bfr[4];
#pragma unroll
    for (int c = 0; c < 4; ++c)
      bfr[c] = load_b_frag(K + qb, ld, c * 16, k0, lane);
#pragma unroll
    for (int c = 0; c < 4; ++c)
      s[c] = wmma_bf16(a, bfr[c], s[c]);
  }

  // Row softmax: each row's 64 values live in one 16-lane half across 4 accs;
  // xor-shuffles with masks 1,2,4,8 reduce each half independently.
  const int ncol = lane & 15;
  const int lh   = lane >> 4;
  float prob[4][8];
#pragma unroll
  for (int r = 0; r < 8; ++r) {
    float m = -3.0e38f;
#pragma unroll
    for (int c = 0; c < 4; ++c) {
      float v = s[c][r] * scale;
      prob[c][r] = v;
      m = fmaxf(m, v);
    }
#pragma unroll
    for (int msk = 1; msk <= 8; msk <<= 1) m = fmaxf(m, __shfl_xor(m, msk));
    float sum = 0.f;
#pragma unroll
    for (int c = 0; c < 4; ++c) {
      float e = __expf(prob[c][r] - m);
      prob[c][r] = e;
      sum += e;
    }
#pragma unroll
    for (int msk = 1; msk <= 8; msk <<= 1) sum += __shfl_xor(sum, msk);
    const float inv = 1.f / sum;
#pragma unroll
    for (int c = 0; c < 4; ++c)
      Pl[wave][lh * 8 + r][c * 16 + ncol] = (bf16)(prob[c][r] * inv);
  }
  __syncthreads();  // also covers the Vt transpose

  // CTX = P * V  (A from LDS P, B from LDS V^T: contiguous over j)
  constexpr int DT = DK / 16;
  v8f o[DT] = {};
#pragma unroll
  for (int j0 = 0; j0 < 64; j0 += 32) {
    v16bf a = load_a_frag(&Pl[wave][0][0], 64, 0, j0, lane);
    v16bf bfr[DT];
#pragma unroll
    for (int d = 0; d < DT; ++d)
      bfr[d] = load_b_frag(&Vt[0][0], 64, d * 16, j0, lane);
#pragma unroll
    for (int d = 0; d < DT; ++d)
      o[d] = wmma_bf16(a, bfr[d], o[d]);
  }
#pragma unroll
  for (int d = 0; d < DT; ++d)
#pragma unroll
    for (int r = 0; r < 8; ++r) {
      const int row = wave * 16 + lh * 8 + r;
      const int col = d * 16 + ncol;
      CTX[ob + (size_t)row * ldo + col] = (bf16)(o[d][r]);
    }
}

// ---- Elementwise / small kernels -------------------------------------------
__global__ void pe_kernel(float* pe) {
  const int i = blockIdx.x * 256 + threadIdx.x;   // T_SEQ*DMODEL
  const int t = i >> 9, d = i & 511;
  const float freq = __expf((float)(d & ~1) * (-9.210340371976184f / 512.0f));
  const float v = (float)t * freq;
  pe[i] = (d & 1) ? __cosf(v) : __sinf(v);
}

__global__ void cvt_f32_bf16(const float* __restrict__ src, bf16* __restrict__ dst, int n) {
  const int i = blockIdx.x * 256 + threadIdx.x;
  if (i < n) dst[i] = (bf16)src[i];
}

// conv1d(kernel=3, SAME) + bias + relu + PE; out token row = t*64+b.
__global__ void conv_embed(const float* __restrict__ x, const float* __restrict__ w,
                           const float* __restrict__ bias, const float* __restrict__ pe,
                           float* __restrict__ outF, bf16* __restrict__ outB, int C) {
  const int i = blockIdx.x * 256 + threadIdx.x;   // NTOK*512
  const int d = i & 511;
  const int b = (i >> 9) & 63;
  const int t = i >> 15;
  const float* xb = x + (size_t)b * C * T_SEQ;
  const float* wd = w + (size_t)d * C * 3;
  float acc = bias[d];
  for (int c = 0; c < C; ++c) {
    const float* xc = xb + c * T_SEQ;
    const float* wc = wd + c * 3;
    if (t > 0)          acc += xc[t - 1] * wc[0];
                        acc += xc[t]     * wc[1];
    if (t < T_SEQ - 1)  acc += xc[t + 1] * wc[2];
  }
  acc = fmaxf(acc, 0.f) + pe[t * DMODEL + d];
  outF[i] = acc;
  outB[i] = (bf16)acc;
}

// h = LN(src)*g+b + res ; emits fp32 + bf16. One block (256 thr) per row.
__global__ void ln_residual(const float* __restrict__ src, const float* __restrict__ res,
                            const float* __restrict__ g, const float* __restrict__ bta,
                            float* __restrict__ outF, bf16* __restrict__ outB) {
  __shared__ float s1[256], s2[256];
  const int row = blockIdx.x, tid = threadIdx.x;
  const size_t base = (size_t)row * DMODEL;
  const float a0 = src[base + tid], a1 = src[base + 256 + tid];
  s1[tid] = a0 + a1;
  s2[tid] = a0 * a0 + a1 * a1;
  __syncthreads();
  for (int s = 128; s > 0; s >>= 1) {
    if (tid < s) { s1[tid] += s1[tid + s]; s2[tid] += s2[tid + s]; }
    __syncthreads();
  }
  const float mean = s1[0] * (1.f / 512.f);
  const float var  = s2[0] * (1.f / 512.f) - mean * mean;
  const float inv  = rsqrtf(var + 1e-5f);
  const float o0 = (a0 - mean) * inv * g[tid]       + bta[tid]       + res[base + tid];
  const float o1 = (a1 - mean) * inv * g[tid + 256] + bta[tid + 256] + res[base + 256 + tid];
  outF[base + tid] = o0;        outB[base + tid] = (bf16)o0;
  outF[base + 256 + tid] = o1;  outB[base + 256 + tid] = (bf16)o1;
}

// cat[b, m*512+d] = xF_m[(T-1)*64 + b, d]  (fp32 into d_out slice + bf16 copy)
__global__ void gather_cat(const float* __restrict__ x0, const float* __restrict__ x1,
                           const float* __restrict__ x2,
                           float* __restrict__ catF, bf16* __restrict__ catB) {
  const int i = blockIdx.x * 256 + threadIdx.x;   // 3*64*512
  const int d = i & 511;
  const int b = (i >> 9) & 63;
  const int m = i >> 15;
  const float* xm = (m == 0) ? x0 : (m == 1) ? x1 : x2;
  const float v = xm[((size_t)(T_SEQ - 1) * 64 + b) * DMODEL + d];
  const size_t o = (size_t)b * 1536 + m * 512 + d;
  catF[o] = v;
  catB[o] = (bf16)v;
}

// y[b] = sigmoid(ctx[b,:] . fc2_w + fc2_b)
__global__ void head_out(const float* __restrict__ ctx, const float* __restrict__ w,
                         const float* __restrict__ b, float* __restrict__ y) {
  const int bb = threadIdx.x;   // 64 threads
  float acc = b[0];
  for (int k = 0; k < DMODEL; ++k) acc += ctx[(size_t)bb * DMODEL + k] * w[k];
  y[bb] = 1.f / (1.f + __expf(-acc));
}

// ---------------------------------------------------------------------------
static inline char* bump(char*& cur, size_t bytes) {
  char* p = cur;
  cur += (bytes + 255) & ~(size_t)255;
  return p;
}

extern "C" void kernel_launch(void* const* d_in, const int* in_sizes, int n_in,
                              void* d_out, int out_size, void* d_ws, size_t ws_size,
                              hipStream_t stream) {
  (void)in_sizes; (void)n_in; (void)out_size; (void)ws_size;
  const float* x[3] = {(const float*)d_in[0], (const float*)d_in[1], (const float*)d_in[2]};
  // d_in[3] = x_len (unused). Params flattened JAX-style (sorted dict keys):
  //  embed:  [b,w] x3 -> idx 0..5
  //  layers: base 6+14*l: b1,b2,bk,bo,bq,bv,ln2_b,ln2_g,w1,w2,wk,wo,wq,wv
  //  out:    base 62: fc1_b,fc1_w,fc2_b,fc2_w,in_b,in_w,out_b,out_w
  const float* Pp[70];
  for (int i = 0; i < 70; ++i) Pp[i] = (const float*)d_in[4 + i];

  // ---- workspace layout ----
  char* cur = (char*)d_ws;
  float* pe = (float*)bump(cur, (size_t)T_SEQ * DMODEL * 4);
  bf16* wB[4][6];   // wq, wk, wv, wo, w1, w2 (bf16)
  for (int l = 0; l < 4; ++l) {
    for (int j = 0; j < 4; ++j) wB[l][j] = (bf16*)bump(cur, (size_t)512 * 512 * 2);
    wB[l][4] = (bf16*)bump(cur, (size_t)2048 * 512 * 2);
    wB[l][5] = (bf16*)bump(cur, (size_t)512 * 2048 * 2);
  }
  bf16* fc1B = (bf16*)bump(cur, (size_t)512 * 1536 * 2);
  bf16* inB  = (bf16*)bump(cur, (size_t)1536 * 512 * 2);
  bf16* owB  = (bf16*)bump(cur, (size_t)512 * 512 * 2);
  float* xFm[3];
  for (int m = 0; m < 3; ++m) xFm[m] = (float*)bump(cur, (size_t)NTOK * DMODEL * 4);
  bf16*  xB   = (bf16*)bump(cur, (size_t)NTOK * DMODEL * 2);
  bf16*  ctxB = (bf16*)bump(cur, (size_t)NTOK * DMODEL * 2);
  float* hF   = (float*)bump(cur, (size_t)NTOK * DMODEL * 4);
  bf16*  hB   = (bf16*)bump(cur, (size_t)NTOK * DMODEL * 2);
  // arena with disjoint lifetimes: [q|k|v] -> [attnF] -> [midB]
  char*  arena = bump(cur, (size_t)NTOK * 2048 * 2);
  bf16*  qB = (bf16*)arena;
  bf16*  kB = qB + (size_t)NTOK * DMODEL;
  bf16*  vB = kB + (size_t)NTOK * DMODEL;
  float* attnF = (float*)arena;
  bf16*  midB  = (bf16*)arena;
  bf16*  catB  = (bf16*)bump(cur, (size_t)64 * 1536 * 2);
  bf16*  h1B   = (bf16*)bump(cur, (size_t)64 * 512 * 2);
  bf16*  qkvB  = (bf16*)bump(cur, (size_t)64 * 1536 * 2);
  bf16*  ctx2B = (bf16*)bump(cur, (size_t)64 * 512 * 2);
  float* ctx2F = (float*)bump(cur, (size_t)64 * 512 * 4);

  // ---- weight fp32 -> bf16 (every call; deterministic) ----
  auto cvt = [&](const float* s, bf16* d, int n) {
    cvt_f32_bf16<<<(n + 255) / 256, 256, 0, stream>>>(s, d, n);
  };
  for (int l = 0; l < 4; ++l) {
    const int base = 6 + 14 * l;
    cvt(Pp[base + 12], wB[l][0], 512 * 512);    // wq
    cvt(Pp[base + 10], wB[l][1], 512 * 512);    // wk
    cvt(Pp[base + 13], wB[l][2], 512 * 512);    // wv
    cvt(Pp[base + 11], wB[l][3], 512 * 512);    // wo
    cvt(Pp[base + 8],  wB[l][4], 2048 * 512);   // w1
    cvt(Pp[base + 9],  wB[l][5], 512 * 2048);   // w2
  }
  cvt(Pp[63], fc1B, 512 * 1536);                // fc1_w
  cvt(Pp[67], inB,  1536 * 512);                // in_w
  cvt(Pp[69], owB,  512 * 512);                 // out_w

  pe_kernel<<<(T_SEQ * DMODEL) / 256, 256, 0, stream>>>(pe);

  // ---- per-modality encoder ----
  const int Cs[3] = {128, 256, 80};
  const dim3 blk(128);
  for (int m = 0; m < 3; ++m) {
    conv_embed<<<(NTOK * DMODEL) / 256, 256, 0, stream>>>(
        x[m], Pp[2 * m + 1], Pp[2 * m], pe, xFm[m], xB, Cs[m]);
    for (int l = 0; l < 4; ++l) {
      const int base = 6 + 14 * l;
      const float *b1 = Pp[base + 0], *b2 = Pp[base + 1], *bk = Pp[base + 2],
                  *bo = Pp[base + 3], *bq = Pp[base + 4], *bv = Pp[base + 5],
                  *lnb = Pp[base + 6], *lng = Pp[base + 7];
      gemm_bf16<false><<<dim3(8, 256), blk, 0, stream>>>(xB, wB[l][0], bq, nullptr, nullptr, qB, NTOK, 512, 512);
      gemm_bf16<false><<<dim3(8, 256), blk, 0, stream>>>(xB, wB[l][1], bk, nullptr, nullptr, kB, NTOK, 512, 512);
      gemm_bf16<false><<<dim3(8, 256), blk, 0, stream>>>(xB, wB[l][2], bv, nullptr, nullptr, vB, NTOK, 512, 512);
      attn_kernel<64><<<T_SEQ * 8, blk, 0, stream>>>(qB, kB, vB, ctxB, 8, 512, 512, 0.125f);
      gemm_bf16<false><<<dim3(8, 256), blk, 0, stream>>>(ctxB, wB[l][3], bo, nullptr, attnF, nullptr, NTOK, 512, 512);
      ln_residual<<<NTOK, 256, 0, stream>>>(attnF, xFm[m], lng, lnb, hF, hB);
      gemm_bf16<true><<<dim3(32, 256), blk, 0, stream>>>(hB, wB[l][4], b1, nullptr, nullptr, midB, NTOK, 2048, 512);
      gemm_bf16<false><<<dim3(8, 256), blk, 0, stream>>>(midB, wB[l][5], b2, hF, xFm[m], xB, NTOK, 512, 2048);
    }
  }

  // ---- output head ----
  float* catF = (float*)d_out + 64;   // second tuple output, written in place
  gather_cat<<<(3 * 64 * 512) / 256, 256, 0, stream>>>(xFm[0], xFm[1], xFm[2], catF, catB);
  gemm_bf16<true ><<<dim3(8, 1),  blk, 0, stream>>>(catB, fc1B, Pp[62], nullptr, nullptr, h1B, 64, 512, 1536);
  gemm_bf16<false><<<dim3(24, 1), blk, 0, stream>>>(h1B, inB, Pp[66], nullptr, nullptr, qkvB, 64, 1536, 512);
  attn_kernel<128><<<4, blk, 0, stream>>>(qkvB, qkvB + 512, qkvB + 1024, ctx2B, 4, 1536, 512, 0.08838834764831845f);
  gemm_bf16<false><<<dim3(8, 1), blk, 0, stream>>>(ctx2B, owB, Pp[68], nullptr, ctx2F, nullptr, 64, 512, 512);
  head_out<<<1, 64, 0, stream>>>(ctx2F, Pp[65], Pp[64], (float*)d_out);
}
